// SCCModule_78022375899286
// MI455X (gfx1250) — matrix operand order repaired and checked by reading, not compiled, not run on hardware
//
#include <hip/hip_runtime.h>
#include <hip/hip_bf16.h>

// ---------------------------------------------------------------------------
// SCC module for MI455X (gfx1250): fused flash-style spatial attentions using
// v_wmma_f32_16x16x32_bf16 with fp32 accumulation.  Q/K operands are N-major
// so every WMMA fragment is two b128 loads; P is repacked through LDS in
// fragment-swizzled order.  Softmax runs in the log2 domain on the raw
// v_exp_f32 / v_rcp_f32 pipes (TRANS ops co-execute with WMMA).
// ---------------------------------------------------------------------------

typedef __attribute__((ext_vector_type(16))) __bf16 v16bf;
typedef __attribute__((ext_vector_type(8)))  float  v8f;
typedef unsigned short u16_t;

#define NN 8192   // d*h*w = 8*32*32
#define CC 64     // channels
#define CR 32     // channels / reduction

// ---- fast math (raw HW transcendentals) ------------------------------------
__device__ __forceinline__ float fexp2(float x) {
#if __has_builtin(__builtin_amdgcn_exp2f)
  return __builtin_amdgcn_exp2f(x);          // v_exp_f32 (2^x), no denorm guard
#else
  return exp2f(x);
#endif
}
__device__ __forceinline__ float frcp(float x) {
#if __has_builtin(__builtin_amdgcn_rcpf)
  return __builtin_amdgcn_rcpf(x);           // v_rcp_f32
#else
  return 1.f / x;
#endif
}

// ---- bf16 helpers ----------------------------------------------------------
__device__ __forceinline__ u16_t f2bf(float f) {
  unsigned int u = __builtin_bit_cast(unsigned int, f);
  u += 0x7FFFu + ((u >> 16) & 1u);            // round-to-nearest-even
  return (u16_t)(u >> 16);
}

struct bf16frag { uint4 lo; uint4 hi; };      // 32 bytes == v16bf

// ---- WMMA wrapper ----------------------------------------------------------
__device__ __forceinline__ v8f wmma_bf16(v16bf a, v16bf b, v8f c) {
  return __builtin_amdgcn_wmma_f32_16x16x32_bf16(false, a, false, b,
                                                 (short)0, c, false, false);
}

// ---- Fragment loaders (wave32; A and B share the contiguous k = 16*(l>>4) + e order) ----------------
// B fragment 32x16 from an N-major matrix T[n][ldk]:
//   element e of lane l is (k = 16*(l>>4) + e, n = colbase + (l&15))
__device__ __forceinline__ v16bf load_fragT_b(const u16_t* __restrict__ T, int ldk,
                                              int colbase, int koff, int lane) {
  const u16_t* p = T + (colbase + (lane & 15)) * ldk + koff + ((lane >> 4) << 4);
  bf16frag r;
  r.lo = *(const uint4*)(p);
  r.hi = *(const uint4*)(p + 8);
  return __builtin_bit_cast(v16bf, r);
}

// A fragment 16x32 from an N-major matrix T[n][ldk]:
//   element e of lane l is (k = 16*(l>>4) + e, m = colbase+(l&15))
__device__ __forceinline__ v16bf load_fragT_a(const u16_t* __restrict__ T, int ldk,
                                              int colbase, int koff, int lane) {
  const u16_t* p = T + (colbase + (lane & 15)) * ldk + koff + ((lane >> 4) << 4);
  bf16frag r;
  r.lo = *(const uint4*)(p);
  r.hi = *(const uint4*)(p + 8);
  return __builtin_bit_cast(v16bf, r);
}

// A fragment 16x32 from a C-major matrix src[row][ldn] (rows = M, k contiguous)
__device__ __forceinline__ v16bf load_frag_a_row(const u16_t* __restrict__ src, int ldn,
                                                 int rowbase, int kbase, int lane) {
  const u16_t* p = src + (rowbase + (lane & 15)) * ldn + kbase + ((lane >> 4) << 4);
  bf16frag r;
  r.lo = *(const uint4*)(p);
  r.hi = *(const uint4*)(p + 8);
  return __builtin_bit_cast(v16bf, r);
}

// B fragment 32x16 from fragment-swizzled LDS (lane's 16 bf16 contiguous)
__device__ __forceinline__ v16bf load_frag_b_lds(const u16_t* lds, int lane) {
  const u16_t* p = lds + lane * 16;
  bf16frag r;
  r.lo = *(const uint4*)(p);
  r.hi = *(const uint4*)(p + 8);
  return __builtin_bit_cast(v16bf, r);
}

// ---------------------------------------------------------------------------
// Small kernels
// ---------------------------------------------------------------------------
__global__ void k_tobf16(const float* __restrict__ in, u16_t* __restrict__ out, int n) {
  int i = blockIdx.x * blockDim.x + threadIdx.x;
  if (i < n) out[i] = f2bf(in[i]);
}

// xrT[n, o] = b_red[o] + sum_c w_red[o,c] * x[c,n]   (N-major bf16)
__global__ void k_xr(const float* __restrict__ x, const float* __restrict__ w,
                     const float* __restrict__ b, u16_t* __restrict__ xrT) {
  const int o = threadIdx.x & (CR - 1);
  const int n = blockIdx.x * 8 + (threadIdx.x >> 5);
  float acc = b[o];
#pragma unroll 8
  for (int c = 0; c < CC; ++c) acc += w[o * CC + c] * x[c * NN + n];
  xrT[n * CR + o] = f2bf(acc);
}

// GAP over spatial + squeeze-excite MLP -> y[64]   (single block, 256 threads)
__global__ void k_gap_mlp(const float* __restrict__ s,
                          const float* __restrict__ w1, const float* __restrict__ b1,
                          const float* __restrict__ w2, const float* __restrict__ b2,
                          float* __restrict__ y) {
  __shared__ float part[256];
  __shared__ float yv[CC];
  __shared__ float hh[8];
  const int tid = threadIdx.x;
  const int c = tid >> 2, t = tid & 3;
  float acc = 0.f;
  for (int n = t; n < NN; n += 4) acc += s[c * NN + n];
  part[tid] = acc;
  __syncthreads();
  if (tid < CC) {
    float tot = part[tid * 4] + part[tid * 4 + 1] + part[tid * 4 + 2] + part[tid * 4 + 3];
    yv[tid] = tot * (1.0f / (float)NN);
  }
  __syncthreads();
  if (tid < 8) {
    float h = b1[tid];
    for (int cc = 0; cc < CC; ++cc) h += w1[tid * CC + cc] * yv[cc];
    hh[tid] = fmaxf(h, 0.f);
  }
  __syncthreads();
  if (tid < CC) {
    float v = b2[tid];
#pragma unroll
    for (int j = 0; j < 8; ++j) v += w2[tid * 8 + j] * hh[j];
    y[tid] = 1.0f / (1.0f + __expf(-v));
  }
}

// cf_in[c,n] = s[c,n] * y[c]
__global__ void k_cfin(const float* __restrict__ s, const float* __restrict__ y,
                       float* __restrict__ cfin) {
  int i = blockIdx.x * blockDim.x + threadIdx.x;
  if (i < CC * NN) cfin[i] = s[i] * y[i / NN];
}

// out[o,n] = b[o] + sum_c W[o,c]*in[c,n]
__global__ void k_proj64(const float* __restrict__ in, const float* __restrict__ w,
                         const float* __restrict__ b, float* __restrict__ outf,
                         u16_t* __restrict__ out_rm, u16_t* __restrict__ outT) {
  const int n = blockIdx.x * blockDim.x + threadIdx.x;
  const int o = blockIdx.y;
  float acc = b[o];
#pragma unroll 8
  for (int c = 0; c < CC; ++c) acc += w[o * CC + c] * in[c * NN + n];
  outf[o * NN + n] = acc;
  const u16_t r = f2bf(acc);
  out_rm[o * NN + n] = r;
  outT[n * CC + o] = r;
}

// M64[c,d] = scale * sum_n sp[c,n]*cf[d,n]
__global__ void k_cc_logits(const float* __restrict__ sp, const float* __restrict__ cf,
                            float* __restrict__ m64, float scale) {
  __shared__ float red[256];
  const int c = blockIdx.x, d = blockIdx.y;
  float acc = 0.f;
  for (int n = threadIdx.x; n < NN; n += 256) acc += sp[c * NN + n] * cf[d * NN + n];
  red[threadIdx.x] = acc;
  __syncthreads();
  for (int off = 128; off > 0; off >>= 1) {
    if (threadIdx.x < off) red[threadIdx.x] += red[threadIdx.x + off];
    __syncthreads();
  }
  if (threadIdx.x == 0) m64[c * CC + d] = red[0] * scale;
}

// row softmax of 64x64 in place  (1 block, 64 threads)
__global__ void k_cc_softmax(float* __restrict__ m64) {
  const int r = threadIdx.x;
  float mx = -1e30f;
  for (int d = 0; d < CC; ++d) mx = fmaxf(mx, m64[r * CC + d]);
  float sm = 0.f;
  for (int d = 0; d < CC; ++d) sm += __expf(m64[r * CC + d] - mx);
  float inv = 1.0f / sm;
  for (int d = 0; d < CC; ++d) m64[r * CC + d] = __expf(m64[r * CC + d] - mx) * inv;
}

// base2[c,n] = cf_in[c,n] + sum_d A[c,d]*cf[d,n]
__global__ void k_s2c_base(const float* __restrict__ a, const float* __restrict__ cf,
                           const float* __restrict__ cfin, float* __restrict__ base) {
  const int n = blockIdx.x * blockDim.x + threadIdx.x;
  const int c = blockIdx.y;
  float acc = cfin[c * NN + n];
#pragma unroll 8
  for (int d = 0; d < CC; ++d) acc += a[c * CC + d] * cf[d * NN + n];
  base[c * NN + n] = acc;
}

// ---------------------------------------------------------------------------
// Flash attention (transposed-normalization form), log2-domain softmax:
//   L2[n,m] = scale2 * sum_k Q[k,n]*K[k,m]   (scale2 = scale*log2 e)
//   P[n,m]  = 2^(L2[n,m]-rowmax2[n]) / rowsum2[n]
//   out[c,m] = base[c,m] + sum_n V[c,n]*P[n,m]
// Each block owns 32 rows (pass1) / 32 columns (pass2).
// ---------------------------------------------------------------------------
template <int KD>
__global__ __launch_bounds__(128)
void k_attn_pass1(const u16_t* __restrict__ QT, const u16_t* __restrict__ KT,
                  float* __restrict__ rowmax, float* __restrict__ rowsum, float scale2) {
  const int lane = threadIdx.x & 31;
  const int wave = threadIdx.x >> 5;   // 0..3
  const int half = lane >> 4;
  const int nbase = blockIdx.x * 32;

  v16bf a[2][2];
#pragma unroll
  for (int rt = 0; rt < 2; ++rt) {
    a[rt][0] = load_fragT_a(QT, KD, nbase + 16 * rt, 0, lane);
    if (KD == 64) a[rt][1] = load_fragT_a(QT, KD, nbase + 16 * rt, 32, lane);
  }

  float mx[16], sm[16];
#pragma unroll
  for (int i = 0; i < 16; ++i) { mx[i] = -1e30f; sm[i] = 0.f; }

  const v8f zero = {0.f, 0.f, 0.f, 0.f, 0.f, 0.f, 0.f, 0.f};
  const int ntiles = NN >> 4;
  for (int mt = wave; mt < ntiles; mt += 4) {
    const int mbase = mt << 4;
    if (mt + 4 < ntiles) __builtin_prefetch(KT + (mbase + 64) * KD, 0, 1);
    v16bf b0 = load_fragT_b(KT, KD, mbase, 0, lane);
    v16bf b1;
    if (KD == 64) b1 = load_fragT_b(KT, KD, mbase, 32, lane);
#pragma unroll
    for (int rt = 0; rt < 2; ++rt) {
      v8f acc = wmma_bf16(a[rt][0], b0, zero);
      if (KD == 64) acc = wmma_bf16(a[rt][1], b1, acc);
#pragma unroll
      for (int v = 0; v < 8; ++v) {
        const int s = rt * 8 + v;
        float l = acc[v] * scale2;
        float nm = fmaxf(mx[s], l);
        sm[s] = sm[s] * fexp2(mx[s] - nm) + fexp2(l - nm);
        mx[s] = nm;
      }
    }
  }

  // reduce (max,sum) across the 16 column lanes of each half-wave
#pragma unroll
  for (int s = 0; s < 16; ++s) {
#pragma unroll
    for (int off = 1; off < 16; off <<= 1) {
      float omx = __shfl_xor(mx[s], off, 32);
      float osm = __shfl_xor(sm[s], off, 32);
      float nm = fmaxf(mx[s], omx);
      sm[s] = sm[s] * fexp2(mx[s] - nm) + osm * fexp2(omx - nm);
      mx[s] = nm;
    }
  }

  __shared__ float smx[4][32];
  __shared__ float ssm[4][32];
  if ((lane & 15) == 0) {
#pragma unroll
    for (int s = 0; s < 16; ++s) {
      const int row = (s >> 3) * 16 + (s & 7) + 8 * half;
      smx[wave][row] = mx[s];
      ssm[wave][row] = sm[s];
    }
  }
  __syncthreads();
  if (threadIdx.x < 32) {
    const int r = threadIdx.x;
    float M = smx[0][r], S = ssm[0][r];
#pragma unroll
    for (int w = 1; w < 4; ++w) {
      float om = smx[w][r], os = ssm[w][r];
      float nm = fmaxf(M, om);
      S = S * fexp2(M - nm) + os * fexp2(om - nm);
      M = nm;
    }
    rowmax[nbase + r] = M;
    rowsum[nbase + r] = S;
  }
}

template <int KD>
__global__ __launch_bounds__(128)
void k_attn_pass2(const u16_t* __restrict__ QT, const u16_t* __restrict__ KT,
                  const u16_t* __restrict__ V,
                  const float* __restrict__ rowmax, const float* __restrict__ rowsum,
                  const float* __restrict__ base, float* __restrict__ out, float scale2) {
  const int lane = threadIdx.x & 31;
  const int wave = threadIdx.x >> 5;   // 0..3
  const int half = lane >> 4;
  const int col  = lane & 15;
  const int mbase = blockIdx.x * 32;

  __shared__ uint4 pbuf4[4][2][64];    // per-wave 32x16 bf16 P tiles (2 m-tiles)
  __shared__ float red[4][CC * 32];    // cross-wave output reduction (32 KB)

  // key fragments for the two fixed m-tiles (load once)
  v16bf bk[2][2];
#pragma unroll
  for (int j = 0; j < 2; ++j) {
    bk[j][0] = load_fragT_b(KT, KD, mbase + 16 * j, 0, lane);
    if (KD == 64) bk[j][1] = load_fragT_b(KT, KD, mbase + 16 * j, 32, lane);
  }

  const v8f zero = {0.f, 0.f, 0.f, 0.f, 0.f, 0.f, 0.f, 0.f};
  v8f oacc[4][2];
#pragma unroll
  for (int vb = 0; vb < 4; ++vb)
#pragma unroll
    for (int j = 0; j < 2; ++j) oacc[vb][j] = zero;

  for (int nc = wave * 32; nc < NN; nc += 128) {
    // two 16-row logit tiles x two m-tiles -> P (bf16) staged to LDS
#pragma unroll
    for (int t = 0; t < 2; ++t) {
      const int nb = nc + t * 16;
      v16bf aq0 = load_fragT_a(QT, KD, nb, 0, lane);
      v16bf aq1;
      if (KD == 64) aq1 = load_fragT_a(QT, KD, nb, 32, lane);
      // 8 consecutive rowmax/rowsum values per half-wave -> wide loads
      const float4* rm4 = (const float4*)&rowmax[nb + (half << 3)];
      const float4* rs4 = (const float4*)&rowsum[nb + (half << 3)];
      float4 m0 = rm4[0], m1 = rm4[1];
      float4 s0 = rs4[0], s1 = rs4[1];
      float rmv[8] = {m0.x, m0.y, m0.z, m0.w, m1.x, m1.y, m1.z, m1.w};
      float rin[8] = {frcp(s0.x), frcp(s0.y), frcp(s0.z), frcp(s0.w),
                      frcp(s1.x), frcp(s1.y), frcp(s1.z), frcp(s1.w)};
#pragma unroll
      for (int j = 0; j < 2; ++j) {
        v8f acc = wmma_bf16(aq0, bk[j][0], zero);
        if (KD == 64) acc = wmma_bf16(aq1, bk[j][1], acc);
        unsigned int pk[4];
#pragma unroll
        for (int v = 0; v < 4; ++v) {
          float p0 = fexp2(__builtin_fmaf(acc[2 * v],     scale2, -rmv[2 * v]))     * rin[2 * v];
          float p1 = fexp2(__builtin_fmaf(acc[2 * v + 1], scale2, -rmv[2 * v + 1])) * rin[2 * v + 1];
          pk[v] = (unsigned int)f2bf(p0) | ((unsigned int)f2bf(p1) << 16);
        }
        // value (k = t*16+v+8*half, m = col) -> frag-swizzled slot; b128 store
        uint4 pv = {pk[0], pk[1], pk[2], pk[3]};
        u16_t* pb = (u16_t*)&pbuf4[wave][j][0];
        *(uint4*)&pb[((t * 16 + col) << 4) + (half << 3)] = pv;
      }
    }
    asm volatile("s_wait_dscnt 0" ::: "memory");   // wave-private LDS RAW

    v16bf bp0 = load_frag_b_lds((const u16_t*)&pbuf4[wave][0][0], lane);
    v16bf bp1 = load_frag_b_lds((const u16_t*)&pbuf4[wave][1][0], lane);
#pragma unroll
    for (int vb = 0; vb < 4; ++vb) {
      v16bf av = load_frag_a_row(V, NN, vb * 16, nc, lane);   // shared by both m-tiles
      oacc[vb][0] = wmma_bf16(av, bp0, oacc[vb][0]);
      oacc[vb][1] = wmma_bf16(av, bp1, oacc[vb][1]);
    }
  }

  // cross-wave reduction of the 64x32 output tile
#pragma unroll
  for (int vb = 0; vb < 4; ++vb)
#pragma unroll
    for (int j = 0; j < 2; ++j)
#pragma unroll
      for (int v = 0; v < 8; ++v)
        red[wave][(vb * 16 + v + 8 * half) * 32 + j * 16 + col] = oacc[vb][j][v];
  __syncthreads();
  for (int i = threadIdx.x; i < CC * 32; i += 128) {
    float s = red[0][i] + red[1][i] + red[2][i] + red[3][i];
    const int c = i >> 5;
    const int m = mbase + (i & 31);
    out[c * NN + m] = base[c * NN + m] + s;
  }
}

// ---------------------------------------------------------------------------
extern "C" void kernel_launch(void* const* d_in, const int* in_sizes, int n_in,
                              void* d_out, int out_size, void* d_ws, size_t ws_size,
                              hipStream_t stream) {
  (void)in_sizes; (void)n_in; (void)out_size; (void)ws_size;
  const float* x     = (const float*)d_in[0];
  const float* w_red = (const float*)d_in[1];
  const float* b_red = (const float*)d_in[2];
  const float* w_fc1 = (const float*)d_in[3];
  const float* b_fc1 = (const float*)d_in[4];
  const float* w_fc2 = (const float*)d_in[5];
  const float* b_fc2 = (const float*)d_in[6];
  const float* w_sp  = (const float*)d_in[7];
  const float* b_sp  = (const float*)d_in[8];
  const float* w_ch  = (const float*)d_in[9];
  const float* b_ch  = (const float*)d_in[10];
  float* out = (float*)d_out;

  char* p = (char*)d_ws;
  auto alloc = [&](size_t bytes) -> void* {
    void* r = (void*)p;
    p += (bytes + 255) & ~(size_t)255;
    return r;
  };
  u16_t* xf_rm  = (u16_t*)alloc((size_t)CC * NN * 2);  // V of stage 1 (C-major)
  u16_t* xr_t   = (u16_t*)alloc((size_t)CR * NN * 2);  // Q/K of stage 1 (N-major)
  float* rm1    = (float*)alloc((size_t)NN * 4);
  float* rs1    = (float*)alloc((size_t)NN * 4);
  float* s_f    = (float*)alloc((size_t)CC * NN * 4);
  float* y      = (float*)alloc((size_t)CC * 4);
  float* cf_in  = (float*)alloc((size_t)CC * NN * 4);
  float* sp_f   = (float*)alloc((size_t)CC * NN * 4);
  u16_t* sp_rm  = (u16_t*)alloc((size_t)CC * NN * 2);  // V of stage 2 (C-major)
  u16_t* sp_t   = (u16_t*)alloc((size_t)CC * NN * 2);  // K of stage 2 (N-major)
  float* cf_f   = (float*)alloc((size_t)CC * NN * 4);
  u16_t* cf_rm  = (u16_t*)alloc((size_t)CC * NN * 2);  // (unused, uniform producer)
  u16_t* cf_t   = (u16_t*)alloc((size_t)CC * NN * 2);  // Q of stage 2 (N-major)
  float* m64    = (float*)alloc((size_t)CC * CC * 4);
  float* base2  = (float*)alloc((size_t)CC * NN * 4);
  float* rm2    = (float*)alloc((size_t)NN * 4);
  float* rs2    = (float*)alloc((size_t)NN * 4);

  const float LOG2E   = 1.4426950408889634f;
  const float scale_s = 0.17677669529663687f * LOG2E;  // (1/sqrt(32)) * log2 e
  const float scale_x = 0.125f;                        // 64^-0.5
  const float scale_x2 = scale_x * LOG2E;

  // 1) bf16 copy of xf (V of stage 1)
  k_tobf16<<<(CC * NN + 255) / 256, 256, 0, stream>>>(x, xf_rm, CC * NN);
  // 2) reduced projection xr (Q/K of stage 1, N-major)
  k_xr<<<NN / 8, 256, 0, stream>>>(x, w_red, b_red, xr_t);
  // 3-4) S-SC fused attention:  s = xf @ softmax(xr^T xr) + xf
  k_attn_pass1<CR><<<NN / 32, 128, 0, stream>>>(xr_t, xr_t, rm1, rs1, scale_s);
  k_attn_pass2<CR><<<NN / 32, 128, 0, stream>>>(xr_t, xr_t, xf_rm, rm1, rs1,
                                                x, s_f, scale_s);
  // 5) squeeze-excite gate
  k_gap_mlp<<<1, 256, 0, stream>>>(s_f, w_fc1, b_fc1, w_fc2, b_fc2, y);
  // 6) channel-scaled feature
  k_cfin<<<(CC * NN + 255) / 256, 256, 0, stream>>>(s_f, y, cf_in);
  // 7-8) 1x1x1 projections (fp32 + bf16 row-major + bf16 N-major)
  k_proj64<<<dim3(NN / 256, CC), 256, 0, stream>>>(cf_in, w_sp, b_sp, sp_f, sp_rm, sp_t);
  k_proj64<<<dim3(NN / 256, CC), 256, 0, stream>>>(cf_in, w_ch, b_ch, cf_f, cf_rm, cf_t);
  // 9-11) S2C (64x64) attention + base = cf_in + out_s2c
  k_cc_logits<<<dim3(CC, CC), 256, 0, stream>>>(sp_f, cf_f, m64, scale_x);
  k_cc_softmax<<<1, CC, 0, stream>>>(m64);
  k_s2c_base<<<dim3(NN / 256, CC), 256, 0, stream>>>(m64, cf_f, cf_in, base2);
  // 12-13) C2S fused attention: out = base + sp @ softmax(cf^T sp)
  k_attn_pass1<CC><<<NN / 32, 128, 0, stream>>>(cf_t, sp_t, rm2, rs2, scale_x2);
  k_attn_pass2<CC><<<NN / 32, 128, 0, stream>>>(cf_t, sp_t, sp_rm, rm2, rs2,
                                                base2, out, scale_x2);
}
